// Model_60808146977104
// MI455X (gfx1250) — matrix operand order repaired and checked by reading, not compiled, not run on hardware
//
#include <hip/hip_runtime.h>
#include <hip/hip_bf16.h>

// ---------------------------------------------------------------------------
// Types for CDNA5 WMMA (wave32)
// ---------------------------------------------------------------------------
typedef __attribute__((ext_vector_type(16))) _Float16 v16h;
typedef __attribute__((ext_vector_type(8)))  _Float16 v8h;
typedef __attribute__((ext_vector_type(8)))  float    v8f;

#define NN 50000
#define EE 800000

// ---------------------------------------------------------------------------
// Generic f16 WMMA GEMM: C[M,Nc] = A[M,K] * Bt[Nc,K]^T (+bias) (fp32 out)
// One wave computes a 16x64 tile (4 accumulators), block = 4 waves.
// M multiple of 16, Nc multiple of 64, K multiple of 32 (all true here).
// ---------------------------------------------------------------------------
__global__ __launch_bounds__(128)
void gemm_wmma_f16(const _Float16* __restrict__ A, const _Float16* __restrict__ Bt,
                   const float* __restrict__ bias, float* __restrict__ C,
                   int M, int Nc, int K)
{
    const int lane   = threadIdx.x & 31;
    const int wave   = threadIdx.x >> 5;
    const int mTile  = blockIdx.y * 4 + wave;
    if (mTile * 16 >= M) return;            // wave-uniform: EXEC stays all-ones
    const int m0     = mTile * 16;
    const int n0     = blockIdx.x * 64;
    const int l15    = lane & 15;
    const int hiHalf = (lane >> 4) & 1;

    v8f acc0 = {}, acc1 = {}, acc2 = {}, acc3 = {};

    // A fragment (16-bit A 16x32): lanes 0-15 -> K{0..15}, lanes 16-31 -> K{16..31}
    const _Float16* arow = A + (size_t)(m0 + l15) * K + hiHalf * 16;
    // B fragment (16-bit B 32x16): lane holds 16 contiguous K for column n = (lane&15)
    const _Float16* b0p = Bt + (size_t)(n0 +  0 + l15) * K + hiHalf * 16;
    const _Float16* b1p = Bt + (size_t)(n0 + 16 + l15) * K + hiHalf * 16;
    const _Float16* b2p = Bt + (size_t)(n0 + 32 + l15) * K + hiHalf * 16;
    const _Float16* b3p = Bt + (size_t)(n0 + 48 + l15) * K + hiHalf * 16;

    for (int k0 = 0; k0 < K; k0 += 32) {
        v8h alo = *(const v8h*)(arow + k0);
        v8h ahi = *(const v8h*)(arow + k0 + 8);
        v16h a  = __builtin_shufflevector(alo, ahi, 0,1,2,3,4,5,6,7,8,9,10,11,12,13,14,15);
        v16h b0 = *(const v16h*)(b0p + k0);
        v16h b1 = *(const v16h*)(b1p + k0);
        v16h b2 = *(const v16h*)(b2p + k0);
        v16h b3 = *(const v16h*)(b3p + k0);
        if (k0 + 32 < K) {                  // gfx1250 global_prefetch_b8 path
            __builtin_prefetch(arow + k0 + 32, 0, 1);
            __builtin_prefetch(b0p + k0 + 32, 0, 1);
        }
        acc0 = __builtin_amdgcn_wmma_f32_16x16x32_f16(false, a, false, b0, (short)0, acc0, false, false);
        acc1 = __builtin_amdgcn_wmma_f32_16x16x32_f16(false, a, false, b1, (short)0, acc1, false, false);
        acc2 = __builtin_amdgcn_wmma_f32_16x16x32_f16(false, a, false, b2, (short)0, acc2, false, false);
        acc3 = __builtin_amdgcn_wmma_f32_16x16x32_f16(false, a, false, b3, (short)0, acc3, false, false);
    }

    // C/D layout: VGPR r -> row m0 + r (lanes 0-15) or m0 + 8 + r (lanes 16-31), col n0 + (lane&15)
    const int mrow = m0 + hiHalf * 8;
    #pragma unroll
    for (int r = 0; r < 8; ++r) {
        float* crow = C + (size_t)(mrow + r) * Nc + n0 + l15;
        float v0 = acc0[r], v1 = acc1[r], v2 = acc2[r], v3 = acc3[r];
        if (bias) {
            v0 += bias[n0 +  0 + l15];
            v1 += bias[n0 + 16 + l15];
            v2 += bias[n0 + 32 + l15];
            v3 += bias[n0 + 48 + l15];
        }
        crow[ 0] = v0; crow[16] = v1; crow[32] = v2; crow[48] = v3;
    }
}

// ---------------------------------------------------------------------------
// Elementwise / graph kernels
// ---------------------------------------------------------------------------
__global__ void f32_to_f16_kernel(const float* __restrict__ s, _Float16* __restrict__ d, long n) {
    long t = (long)blockIdx.x * blockDim.x + threadIdx.x;
    if (t < n) d[t] = (_Float16)s[t];
}

__global__ void fill_kernel(float* __restrict__ p, float v, long n) {
    long t = (long)blockIdx.x * blockDim.x + threadIdx.x;
    if (t < n) p[t] = v;
}

// es[h,n] = dot(Z[n, h*128 : h*128+128], a_s[h]);  ed likewise
__global__ void attn_coeff_kernel(const float* __restrict__ Z, const float* __restrict__ a_s,
                                  const float* __restrict__ a_d,
                                  float* __restrict__ es, float* __restrict__ ed, int N) {
    int t = blockIdx.x * blockDim.x + threadIdx.x;
    if (t >= 2 * N) return;
    int h = t & 1, n = t >> 1;
    const float* zr = Z + (size_t)n * 256 + h * 128;
    const float* as = a_s + h * 128;
    const float* ad = a_d + h * 128;
    float s1 = 0.f, s2 = 0.f;
    #pragma unroll 4
    for (int o = 0; o < 128; ++o) { float z = zr[o]; s1 += z * as[o]; s2 += z * ad[o]; }
    es[(size_t)h * N + n] = s1;
    ed[(size_t)h * N + n] = s2;
}

__device__ inline float lrelu(float x) { return x > 0.f ? x : 0.2f * x; }

__device__ inline void atomicMaxFloat(float* addr, float val) {
    unsigned int* ua = (unsigned int*)addr;
    unsigned int old = *ua;
    for (;;) {
        float f = __uint_as_float(old);
        if (f >= val) break;
        unsigned int assumed = old;
        old = atomicCAS(ua, assumed, __float_as_uint(val));
        if (old == assumed) break;
    }
}

__global__ void edge_max_kernel(const int* __restrict__ src, const int* __restrict__ dst,
                                const float* __restrict__ es, const float* __restrict__ ed,
                                float* __restrict__ mx, int N, int E) {
    int e = blockIdx.x * blockDim.x + threadIdx.x;
    if (e >= E) return;
    int s = src[e], d = dst[e];
    #pragma unroll
    for (int h = 0; h < 2; ++h) {
        float v = lrelu(es[(size_t)h * N + s] + ed[(size_t)h * N + d]);
        atomicMaxFloat(&mx[(size_t)h * N + d], v);
    }
}

__global__ void edge_exp_kernel(const int* __restrict__ src, const int* __restrict__ dst,
                                const float* __restrict__ es, const float* __restrict__ ed,
                                const float* __restrict__ mx, float* __restrict__ den,
                                float* __restrict__ exb, int N, int E) {
    int e = blockIdx.x * blockDim.x + threadIdx.x;
    if (e >= E) return;
    int s = src[e], d = dst[e];
    #pragma unroll
    for (int h = 0; h < 2; ++h) {
        float v  = lrelu(es[(size_t)h * N + s] + ed[(size_t)h * N + d]);
        float ex = __expf(v - mx[(size_t)h * N + d]);
        exb[(size_t)h * E + e] = ex;
        atomicAdd(&den[(size_t)h * N + d], ex);
    }
}

// 32 lanes per (edge, head); each lane moves 4 channels (float4)
__global__ void edge_aggregate_kernel(const int* __restrict__ src, const int* __restrict__ dst,
                                      const float* __restrict__ exb, const float* __restrict__ den,
                                      const float* __restrict__ Z, float* __restrict__ agg,
                                      int N, int E) {
    long t = (long)blockIdx.x * blockDim.x + threadIdx.x;
    if (t >= (long)E * 64) return;
    int  lane = (int)(t & 31);
    long eh   = t >> 5;
    int  h    = (int)(eh & 1);
    long e    = eh >> 1;
    int  s = src[e], d = dst[e];
    float a = exb[(size_t)h * E + e] / den[(size_t)h * N + d];
    const float4 z = *(const float4*)(Z + (size_t)s * 256 + h * 128 + lane * 4);
    float* o = agg + (size_t)d * 256 + h * 128 + lane * 4;
    atomicAdd(o + 0, a * z.x);
    atomicAdd(o + 1, a * z.y);
    atomicAdd(o + 2, a * z.z);
    atomicAdd(o + 3, a * z.w);
}

__global__ void relu_f16_kernel(const float* __restrict__ in, _Float16* __restrict__ out, long n) {
    long t = (long)blockIdx.x * blockDim.x + threadIdx.x;
    if (t < n) { float v = in[t]; out[t] = (_Float16)(v > 0.f ? v : 0.f); }
}

// GRU cell. gi/gh are [N,768] pre-activation (+bias already fused by GEMM).
// gh==null  -> h_prev = 0 step: gh terms are just bh (broadcast).
// hprev may alias hsum; output hsum[t] = hprev + h_new (accumulates the JKN layer sum).
__global__ void gru_cell_kernel(const float* __restrict__ gi, const float* __restrict__ gh,
                                const float* __restrict__ bh, const float* __restrict__ hprev,
                                float* __restrict__ hsum, _Float16* __restrict__ h16, int N) {
    long t = (long)blockIdx.x * blockDim.x + threadIdx.x;
    if (t >= (long)N * 256) return;
    long n = t >> 8;
    int  j = (int)(t & 255);
    const float* gir = gi + (size_t)n * 768;
    float ir = gir[j], iz = gir[256 + j], in = gir[512 + j];
    float hr, hz, hn;
    if (gh) { const float* ghr = gh + (size_t)n * 768; hr = ghr[j]; hz = ghr[256 + j]; hn = ghr[512 + j]; }
    else    { hr = bh[j]; hz = bh[256 + j]; hn = bh[512 + j]; }
    float hp = hprev ? hprev[t] : 0.f;
    float r  = 1.f / (1.f + __expf(-(ir + hr)));
    float zt = 1.f / (1.f + __expf(-(iz + hz)));
    float nn = tanhf(in + r * hn);
    float hnew = (1.f - zt) * nn + zt * hp;
    hsum[t] = hp + hnew;                 // step1: hp==0 -> init; step2: layer sum
    if (h16) h16[t] = (_Float16)hnew;
}

// h_N = [sumf | sumb] as f16 rows of width 512 (JKN softmax over singleton == 1.0)
__global__ void concat_f16_kernel(const float* __restrict__ sf, const float* __restrict__ sb,
                                  _Float16* __restrict__ hn, int N) {
    long t = (long)blockIdx.x * blockDim.x + threadIdx.x;
    if (t >= (long)N * 512) return;
    long n = t >> 9;
    int  j = (int)(t & 511);
    float v = (j < 256) ? sf[(size_t)n * 256 + j] : sb[(size_t)n * 256 + (j - 256)];
    hn[t] = (_Float16)v;
}

// ---------------------------------------------------------------------------
// Host orchestration
// ---------------------------------------------------------------------------
extern "C" void kernel_launch(void* const* d_in, const int* in_sizes, int n_in,
                              void* d_out, int out_size, void* d_ws, size_t ws_size,
                              hipStream_t stream) {
    (void)in_sizes; (void)n_in; (void)out_size; (void)ws_size;
    const float* in_feat = (const float*)d_in[0];
    const int*   src     = (const int*)  d_in[1];
    const int*   dst     = (const int*)  d_in[2];
    const float* W1   = (const float*)d_in[3];
    const float* as1  = (const float*)d_in[4];
    const float* ad1  = (const float*)d_in[5];
    const float* W2   = (const float*)d_in[6];
    const float* as2  = (const float*)d_in[7];
    const float* ad2  = (const float*)d_in[8];
    const float* wi_f = (const float*)d_in[9];
    const float* wh_f = (const float*)d_in[10];
    const float* bi_f = (const float*)d_in[11];
    const float* bh_f = (const float*)d_in[12];
    const float* wi_b = (const float*)d_in[13];
    const float* wh_b = (const float*)d_in[14];
    const float* bi_b = (const float*)d_in[15];
    const float* bh_b = (const float*)d_in[16];
    /* d_in[17] = s_w: softmax over a singleton axis -> weight identically 1.0 (dead) */
    const float* lin_w = (const float*)d_in[18];
    float* out = (float*)d_out;

    const int N = NN, E = EE;

    // ---- workspace carving (phase-aliased regions) ----
    char*  ws  = (char*)d_ws;
    size_t off = 0;
    auto carve = [&](size_t bytes) -> char* {
        char* p = ws + off; off += (bytes + 255) & ~(size_t)255; return p;
    };
    float*    BIG  = (float*)   carve((size_t)2 * N * 768 * sizeof(float)); // 307 MB arena
    float*    SUMF = (float*)   carve((size_t)N * 256 * sizeof(float));
    float*    SUMB = (float*)   carve((size_t)N * 256 * sizeof(float));
    _Float16* F0   = (_Float16*)carve((size_t)N * 256 * 2);  // x16 -> f0_16 -> b1_16
    _Float16* H1   = (_Float16*)carve((size_t)N * 256 * 2);
    _Float16* H2   = (_Float16*)carve((size_t)N * 256 * 2);
    _Float16* W1h  = (_Float16*)carve((size_t)65536  * 2);
    _Float16* W2h  = (_Float16*)carve((size_t)65536  * 2);
    _Float16* WiFh = (_Float16*)carve((size_t)196608 * 2);
    _Float16* WhFh = (_Float16*)carve((size_t)196608 * 2);
    _Float16* WiBh = (_Float16*)carve((size_t)196608 * 2);
    _Float16* WhBh = (_Float16*)carve((size_t)196608 * 2);
    _Float16* LINh = (_Float16*)carve((size_t)131072 * 2);
    // aliases into BIG (GAT phase)
    float* Z   = BIG;
    float* AGG = BIG + (size_t)N * 256;
    float* EXB = BIG + (size_t)2 * N * 256;
    float* ES  = EXB + (size_t)2 * E;
    float* ED  = ES + 2 * N;
    float* MX  = ED + 2 * N;
    float* DEN = MX + 2 * N;
    // aliases into BIG (GRU phase: Z/AGG/EXB/attn dead)
    float* G1 = BIG;
    float* G2 = BIG + (size_t)N * 768;
    // alias into BIG (final phase: G1/G2 dead)
    _Float16* HN = (_Float16*)BIG;

    auto blks = [](long n) { return (unsigned)((n + 255) / 256); };
    auto cvt  = [&](const float* s, _Float16* d, long n) {
        f32_to_f16_kernel<<<blks(n), 256, 0, stream>>>(s, d, n);
    };
    auto gemm = [&](const _Float16* A, const _Float16* Bt, const float* bias,
                    float* C, int M, int Nc, int K) {
        dim3 grid((unsigned)(Nc / 64), (unsigned)((M / 16 + 3) / 4));
        gemm_wmma_f16<<<grid, 128, 0, stream>>>(A, Bt, bias, C, M, Nc, K);
    };

    // ---- weights & input to f16 ----
    cvt(W1, W1h, 65536);   cvt(W2, W2h, 65536);
    cvt(wi_f, WiFh, 196608); cvt(wh_f, WhFh, 196608);
    cvt(wi_b, WiBh, 196608); cvt(wh_b, WhBh, 196608);
    cvt(lin_w, LINh, 131072);
    cvt(in_feat, F0, (long)N * 256);

    // ---- one GAT layer ----
    auto gat = [&](const _Float16* X16, const _Float16* Wh16,
                   const float* a_s, const float* a_d, _Float16* Hout) {
        gemm(X16, Wh16, nullptr, Z, N, 256, 256);                              // z = x @ W^T
        attn_coeff_kernel<<<blks(2L * N), 256, 0, stream>>>(Z, a_s, a_d, ES, ED, N);
        fill_kernel<<<blks(2L * N), 256, 0, stream>>>(MX, -3.0e38f, 2L * N);
        fill_kernel<<<blks(2L * N), 256, 0, stream>>>(DEN, 0.f, 2L * N);
        fill_kernel<<<blks((long)N * 256), 256, 0, stream>>>(AGG, 0.f, (long)N * 256);
        edge_max_kernel<<<blks(E), 256, 0, stream>>>(src, dst, ES, ED, MX, N, E);
        edge_exp_kernel<<<blks(E), 256, 0, stream>>>(src, dst, ES, ED, MX, DEN, EXB, N, E);
        edge_aggregate_kernel<<<blks((long)E * 64), 256, 0, stream>>>(src, dst, EXB, DEN, Z, AGG, N, E);
        relu_f16_kernel<<<blks((long)N * 256), 256, 0, stream>>>(AGG, Hout, (long)N * 256);
    };

    gat(F0, W1h, as1, ad1, H1);    // layer 1 (F0 = in_feat f16)
    gat(H1, W2h, as2, ad2, H2);    // layer 2

    const long NC = (long)N * 256;

    // ---- bidirectional GRU over the 2 layer outputs (h0 = 0) ----
    // forward: f0 = cell(h1, h=0); f1 = cell(h2, h=f0);  SUMF = f0 + f1
    gemm(H1, WiFh, bi_f, G1, N, 768, 256);
    gru_cell_kernel<<<blks(NC), 256, 0, stream>>>(G1, nullptr, bh_f, nullptr, SUMF, F0, N);
    gemm(H2, WiFh, bi_f, G1, N, 768, 256);
    gemm(F0, WhFh, bh_f, G2, N, 768, 256);
    gru_cell_kernel<<<blks(NC), 256, 0, stream>>>(G1, G2, bh_f, SUMF, SUMF, nullptr, N);
    // backward (reverse scan): b1 = cell(h2, h=0); b0 = cell(h1, h=b1);  SUMB = b0 + b1
    gemm(H2, WiBh, bi_b, G1, N, 768, 256);
    gru_cell_kernel<<<blks(NC), 256, 0, stream>>>(G1, nullptr, bh_b, nullptr, SUMB, F0, N);
    gemm(H1, WiBh, bi_b, G1, N, 768, 256);
    gemm(F0, WhBh, bh_b, G2, N, 768, 256);
    gru_cell_kernel<<<blks(NC), 256, 0, stream>>>(G1, G2, bh_b, SUMB, SUMB, nullptr, N);

    // ---- JKN (weights == 1) + output linear ----
    concat_f16_kernel<<<blks((long)N * 512), 256, 0, stream>>>(SUMF, SUMB, HN, N);
    gemm(HN, LINh, nullptr, out, N, 256, 512);
}
